// LSTM_40072044871967
// MI455X (gfx1250) — matrix operand order repaired and checked by reading, not compiled, not run on hardware
//
#include <hip/hip_runtime.h>
#include <cstdint>

// Problem constants (match reference)
#define B_SZ   1024
#define T_SZ   128
#define I_SZ   1024
#define H_SZ   1024
#define N4     4096      // 4*H (gates i,f,g,o)
#define K_TOT  2048      // I + H  (A = [x_t | h])
#define NK     64        // K_TOT / KS
#define BM     128       // block tile rows (batch)
#define BN     64        // block tile cols per gate
#define KS     32        // K chunk per WMMA
#define ASTR   20        // dwords per LDS A-row (16 data + 4 pad; 80B = 5*16B aligned)
#define BSTR   136       // dwords per LDS B-row (128 data + 8 pad; 544B = 34*16B aligned)

typedef __attribute__((ext_vector_type(16))) __bf16    v16bf;
typedef __attribute__((ext_vector_type(8)))  float     v8f;
typedef __attribute__((ext_vector_type(4)))  uint32_t  u32x4;
typedef __attribute__((ext_vector_type(4)))  float     f32x4;

union Frag16 { u32x4 q[2]; v16bf v; };

__device__ __forceinline__ uint16_t f2bf(float f) {
  uint32_t u = __float_as_uint(f);
  u += 0x7FFFu + ((u >> 16) & 1u);          // round-to-nearest-even
  return (uint16_t)(u >> 16);
}
__device__ __forceinline__ uint32_t pack2(float a, float b) {
  return (uint32_t)f2bf(a) | ((uint32_t)f2bf(b) << 16);
}
__device__ __forceinline__ float sigmoidf_(float x) {
  return 1.0f / (1.0f + __expf(-x));
}

// LDS byte address for async global->LDS DMA: flat-addr[31:0] is the LDS address.
__device__ __forceinline__ uint32_t lds_addr32(const void* p) {
  return (uint32_t)(uintptr_t)p;
}
// Async copy 16B global -> LDS (tracked by ASYNCcnt, bypasses VGPRs).
template <int OFF>
__device__ __forceinline__ void async_ld_b128(uint32_t laddr, const void* g) {
  asm volatile("global_load_async_to_lds_b128 %0, %1, off offset:%2"
               :: "v"(laddr), "v"(g), "i"(OFF) : "memory");
}
__device__ __forceinline__ void wait_async0() {
  asm volatile("s_wait_asynccnt 0x0" ::: "memory");
}

// ---------------------------------------------------------------------------
// Pack W_cat (K_TOT x N4) bf16: rows 0..1023 = Wx (gate-concat), 1024.. = Wh
// ---------------------------------------------------------------------------
__global__ void pack_weights(const float* __restrict__ wx0, const float* __restrict__ wx1,
                             const float* __restrict__ wx2, const float* __restrict__ wx3,
                             const float* __restrict__ wh0, const float* __restrict__ wh1,
                             const float* __restrict__ wh2, const float* __restrict__ wh3,
                             uint16_t* __restrict__ Wcat) {
  int gid = blockIdx.x * blockDim.x + threadIdx.x;   // 0 .. 2048*4096-1
  int k = gid >> 12;
  int n = gid & (N4 - 1);
  int g = n >> 10;
  int col = n & (H_SZ - 1);
  const float* s;
  if (k < I_SZ) s = (g == 0) ? wx0 : (g == 1) ? wx1 : (g == 2) ? wx2 : wx3;
  else          s = (g == 0) ? wh0 : (g == 1) ? wh1 : (g == 2) ? wh2 : wh3;
  int kk = (k < I_SZ) ? k : (k - I_SZ);
  Wcat[(size_t)k * N4 + n] = f2bf(s[(size_t)kk * H_SZ + col]);
}

__global__ void pack_bias(const float* __restrict__ bx0, const float* __restrict__ bx1,
                          const float* __restrict__ bx2, const float* __restrict__ bx3,
                          const float* __restrict__ bh0, const float* __restrict__ bh1,
                          const float* __restrict__ bh2, const float* __restrict__ bh3,
                          float* __restrict__ bias) {
  int n = blockIdx.x * blockDim.x + threadIdx.x;     // 0..4095
  int g = n >> 10;
  int col = n & (H_SZ - 1);
  const float* bx = (g == 0) ? bx0 : (g == 1) ? bx1 : (g == 2) ? bx2 : bx3;
  const float* bh = (g == 0) ? bh0 : (g == 1) ? bh1 : (g == 2) ? bh2 : bh3;
  bias[n] = bx[col] + bh[col];
}

__global__ void init_state(uint32_t* __restrict__ h0_bits, uint32_t* __restrict__ c_bits) {
  int i = blockIdx.x * blockDim.x + threadIdx.x;     // B*H threads
  c_bits[i] = 0u;                                    // c = 0.0f
  if (i < (B_SZ * H_SZ) / 2) h0_bits[i] = 0u;        // h (bf16 pairs) = 0
}

// ---------------------------------------------------------------------------
// One LSTM step: gates = [x_t|h] @ Wcat + b, fused activations + c/h update.
// Block = 256 threads (8 wave32), tile: 128 rows x 64 cols, all 4 gates.
// LDS double-buffered A and B tiles; B (and A h-phase) staged by async
// global->LDS DMA, x-phase converted f32->bf16 through VGPRs.
// ---------------------------------------------------------------------------
__global__ __launch_bounds__(256)
void lstm_step(const float* __restrict__ x,
               const uint16_t* __restrict__ Wcat,
               const float* __restrict__ bias,
               const uint16_t* __restrict__ hprev,
               uint16_t* __restrict__ hnext,
               float* __restrict__ cbuf,
               int t,
               float* __restrict__ hOut,
               float* __restrict__ cOut) {
  __shared__ uint32_t ldsA[2][BM * ASTR];   // A tile: 128 x 32 bf16 (padded rows)
  __shared__ uint32_t ldsB[2][KS * BSTR];   // B tile: 32 x 256 bf16 (padded rows)

  const int tid  = threadIdx.x;
  const int lane = tid & 31;
  const int wave = tid >> 5;       // 0..7
  const int wm   = wave >> 1;      // 0..3 -> rows wm*32..+31 (2 m-tiles)
  const int wn   = wave & 1;       // 0..1 -> cols wn*32..+31 per gate (2 n-tiles)
  const int n0   = blockIdx.x * BN;   // gate-local column base
  const int m0   = blockIdx.y * BM;   // batch row base

  v8f acc[4][2][2];                // [gate][mtile][ntile]
  #pragma unroll
  for (int g = 0; g < 4; ++g)
    #pragma unroll
    for (int mt = 0; mt < 2; ++mt)
      #pragma unroll
      for (int nt = 0; nt < 2; ++nt) {
        v8f z = {};
        acc[g][mt][nt] = z;
      }

  // ---- A staging geometry: 2 threads per row, 16 K-columns each
  const int srow = tid >> 1;            // 0..127
  const int scol = (tid & 1) * 16;      // 0 or 16
  const float*    xbase = x + (size_t)(m0 + srow) * (T_SZ * I_SZ) + (size_t)t * I_SZ + scol;
  const uint16_t* hbase = hprev + (size_t)(m0 + srow) * H_SZ + scol;
  const int aoff = srow * ASTR + (scol >> 1);

  // ---- B staging geometry: 8 threads per row, 32 columns (64B) each
  const int brow = tid >> 3;            // 0..31
  const int bc0  = (tid & 7) * 32;      // tile-col base (never crosses a gate's 64-col span)
  const size_t bgoff = (size_t)((bc0 >> 6) * H_SZ + n0 + (bc0 & 63));
  const int boff = brow * BSTR + (bc0 >> 1);

  const uint32_t bl[2] = { lds_addr32(&ldsB[0][boff]), lds_addr32(&ldsB[1][boff]) };
  const uint32_t al[2] = { lds_addr32(&ldsA[0][aoff]), lds_addr32(&ldsA[1][aoff]) };

  // ---- prologue: stage k-chunk 0 (x phase) into buffer 0
  {
    const f32x4* src = reinterpret_cast<const f32x4*>(xbase);
    uint32_t* ad = &ldsA[0][aoff];
    #pragma unroll
    for (int j = 0; j < 4; ++j) {
      f32x4 f = src[j];
      ad[2 * j + 0] = pack2(f.x, f.y);
      ad[2 * j + 1] = pack2(f.z, f.w);
    }
    const uint16_t* bg = Wcat + (size_t)brow * N4 + bgoff;
    async_ld_b128<0>(bl[0], bg);
    async_ld_b128<16>(bl[0], bg);
    async_ld_b128<32>(bl[0], bg);
    async_ld_b128<48>(bl[0], bg);
    wait_async0();
  }
  __syncthreads();

  f32x4 xr[4];   // next-chunk x staging regs (x phase only)

  for (int kk = 0; kk < NK; ++kk) {
    const int cur = kk & 1;
    const int nb  = cur ^ 1;
    const int k0n = (kk + 1) * KS;
    const bool hasNext = (kk + 1) < NK;
    const bool nextIsX = k0n < I_SZ;

    // ---- issue next chunk's loads; async DMA lands directly in buffer nb
    // (nb was last read before the previous barrier -> safe to overwrite)
    if (hasNext) {
      const uint16_t* bg = Wcat + (size_t)(k0n + brow) * N4 + bgoff;
      async_ld_b128<0>(bl[nb], bg);
      async_ld_b128<16>(bl[nb], bg);
      async_ld_b128<32>(bl[nb], bg);
      async_ld_b128<48>(bl[nb], bg);
      if (nextIsX) {
        const f32x4* src = reinterpret_cast<const f32x4*>(xbase + k0n);
        #pragma unroll
        for (int j = 0; j < 4; ++j) xr[j] = src[j];
      } else {
        const uint16_t* hg = hbase + (k0n - I_SZ);
        async_ld_b128<0>(al[nb], hg);
        async_ld_b128<16>(al[nb], hg);
      }
    }

    // ---- A fragments from LDS per ISA 16-bit A layout:
    // lanes 0-15: K pairs {0..7,16..23}; lanes 16-31: {8..15,24..31}
    Frag16 afrag[2];
    const int off = (lane >> 4) * 4;
    #pragma unroll
    for (int mt = 0; mt < 2; ++mt) {
      const uint32_t* rowp = &ldsA[cur][(wm * 32 + mt * 16 + (lane & 15)) * ASTR];
      afrag[mt].q[0] = *reinterpret_cast<const u32x4*>(rowp + off);
      afrag[mt].q[1] = *reinterpret_cast<const u32x4*>(rowp + 8 + off);
    }

    // ---- B fragments from LDS (lane = N column, K in A's table order) + WMMA
    #pragma unroll
    for (int g = 0; g < 4; ++g) {
      #pragma unroll
      for (int nt = 0; nt < 2; ++nt) {
        Frag16 bfrag;
        const uint16_t* bp = reinterpret_cast<const uint16_t*>(&ldsB[cur][0]) + (g * 64 + wn * 32 + nt * 16 + (lane & 15));
        #pragma unroll
        for (int s = 0; s < 16; ++s) {
          const int kr = (s < 8) ? ((lane >> 4) * 8 + s) : (16 + (lane >> 4) * 8 + (s - 8));
          bfrag.v[s] = __builtin_bit_cast(__bf16, bp[kr * 2 * BSTR]);
        }
        #pragma unroll
        for (int mt = 0; mt < 2; ++mt)
          acc[g][mt][nt] = __builtin_amdgcn_wmma_f32_16x16x32_bf16(
              false, afrag[mt].v, false, bfrag.v, (short)0, acc[g][mt][nt], false, false);
      }
    }

    // ---- finish staging next chunk
    if (hasNext) {
      if (nextIsX) {
        uint32_t* ad = &ldsA[nb][aoff];
        #pragma unroll
        for (int j = 0; j < 4; ++j) {
          ad[2 * j + 0] = pack2(xr[j].x, xr[j].y);
          ad[2 * j + 1] = pack2(xr[j].z, xr[j].w);
        }
      }
      wait_async0();   // async DMA into buffer nb complete (this wave's issues)
    }
    __syncthreads();
  }

  // ---- fused LSTM epilogue. C/D layout: VGPR v -> M = v + 8*(lane>=16), N = lane&15
  const int half = lane >> 4;
  const int nloc = lane & 15;
  #pragma unroll
  for (int nt = 0; nt < 2; ++nt) {
    const int nH = n0 + wn * 32 + nt * 16 + nloc;     // 0..1023
    const float bi = bias[nH];
    const float bf = bias[H_SZ + nH];
    const float bg = bias[2 * H_SZ + nH];
    const float bo = bias[3 * H_SZ + nH];
    #pragma unroll
    for (int mt = 0; mt < 2; ++mt) {
      #pragma unroll
      for (int v = 0; v < 8; ++v) {
        const int mG = m0 + wm * 32 + mt * 16 + v + 8 * half;
        const size_t idx = (size_t)mG * H_SZ + nH;
        float gi = sigmoidf_(acc[0][mt][nt][v] + bi);
        float gf = sigmoidf_(acc[1][mt][nt][v] + bf);
        float gg = tanhf   (acc[2][mt][nt][v] + bg);
        float go = sigmoidf_(acc[3][mt][nt][v] + bo);
        float cn = gf * cbuf[idx] + gi * gg;
        float hn = go * tanhf(cn);
        cbuf[idx]  = cn;
        hnext[idx] = f2bf(hn);
        if (hOut) { hOut[idx] = hn; cOut[idx] = cn; }
      }
    }
  }
}

// ---------------------------------------------------------------------------
extern "C" void kernel_launch(void* const* d_in, const int* in_sizes, int n_in,
                              void* d_out, int out_size, void* d_ws, size_t ws_size,
                              hipStream_t stream) {
  (void)in_sizes; (void)n_in; (void)out_size; (void)ws_size;
  const float* x   = (const float*)d_in[0];
  const float* Wii = (const float*)d_in[1];
  const float* bii = (const float*)d_in[2];
  const float* Whi = (const float*)d_in[3];
  const float* bhi = (const float*)d_in[4];
  const float* Wif = (const float*)d_in[5];
  const float* bif = (const float*)d_in[6];
  const float* Whf = (const float*)d_in[7];
  const float* bhf = (const float*)d_in[8];
  const float* Wig = (const float*)d_in[9];
  const float* big = (const float*)d_in[10];
  const float* Whg = (const float*)d_in[11];
  const float* bhg = (const float*)d_in[12];
  const float* Wio = (const float*)d_in[13];
  const float* bio = (const float*)d_in[14];
  const float* Who = (const float*)d_in[15];
  const float* bho = (const float*)d_in[16];

  // Workspace layout (~25 MB total)
  char* ws = (char*)d_ws;
  uint16_t* Wcat = (uint16_t*)ws;
  size_t off = (size_t)K_TOT * N4 * sizeof(uint16_t);          // 16 MB
  float* bias = (float*)(ws + off); off += (size_t)N4 * sizeof(float);
  uint16_t* hA = (uint16_t*)(ws + off); off += (size_t)B_SZ * H_SZ * sizeof(uint16_t);
  uint16_t* hB = (uint16_t*)(ws + off); off += (size_t)B_SZ * H_SZ * sizeof(uint16_t);
  float* cbuf = (float*)(ws + off);

  pack_weights<<<(K_TOT * N4) / 256, 256, 0, stream>>>(Wii, Wif, Wig, Wio,
                                                       Whi, Whf, Whg, Who, Wcat);
  pack_bias<<<N4 / 256, 256, 0, stream>>>(bii, bif, big, bio, bhi, bhf, bhg, bho, bias);
  init_state<<<(B_SZ * H_SZ) / 256, 256, 0, stream>>>((uint32_t*)hA, (uint32_t*)cbuf);

  dim3 grid(H_SZ / BN, B_SZ / BM);   // 16 x 8 = 128 blocks
  for (int t = 0; t < T_SZ; ++t) {
    const uint16_t* hp = (t & 1) ? hB : hA;
    uint16_t*       hn = (t & 1) ? hA : hB;
    float* hOut = (t == T_SZ - 1) ? (float*)d_out : nullptr;
    float* cOut = (t == T_SZ - 1) ? ((float*)d_out + (size_t)B_SZ * H_SZ) : nullptr;
    lstm_step<<<grid, 256, 0, stream>>>(x, Wcat, bias, hp, hn, cbuf, t, hOut, cOut);
  }
}
